// MultiHeadAttention_83605833384184
// MI455X (gfx1250) — matrix operand order repaired and checked by reading, not compiled, not run on hardware
//
#include <hip/hip_runtime.h>

// ---------------------------------------------------------------------------
// MHA for MI455X (gfx1250, wave32).
// Pipeline:
//   1) convert q -> bf16;  transpose+convert Wq/Wk/Wv/Wo -> bf16 Wt[n][k]
//   2) 3x async-DMA double-buffered bf16 WMMA GEMMs  -> Qh (pre-scaled by
//      1/sqrt(64)), Kh ([b,h,s,d]), Vt ([b,h,d,s])
//   3) flash attention (WMMA QK^T + PV, online softmax)  -> Cx bf16 [4096,1024]
//   4) async-DMA WMMA GEMM Cx@Wo + bo -> f32 out
// GEMM tiles are filled with global_load_async_to_lds_b128 (ASYNCcnt) so the
// inner loop is pure ds_load + v_wmma with DMA overlapped via double buffering.
// The softmax scale is folded into Q (exact power-of-two in bf16), removing
// the WMMA->VALU hazard chain from the attention inner loop.
// ---------------------------------------------------------------------------

typedef __attribute__((ext_vector_type(16))) __bf16 v16bf;
typedef __attribute__((ext_vector_type(8)))  float  v8f;

union Frag16 {
  v16bf v;
  unsigned int u[8];
  uint4 q[2];
};

__device__ __forceinline__ unsigned short f2bf(float f) {
  unsigned int x = __float_as_uint(f);
  x += 0x7FFFu + ((x >> 16) & 1u);  // round-to-nearest-even
  return (unsigned short)(x >> 16);
}

__device__ __forceinline__ v8f vzero8() {
  v8f r;
#pragma unroll
  for (int i = 0; i < 8; ++i) r[i] = 0.0f;
  return r;
}

__device__ __forceinline__ v8f wmma_bf16(v16bf a, v16bf b, v8f c) {
  return __builtin_amdgcn_wmma_f32_16x16x32_bf16(false, a, false, b, (short)0, c,
                                                 false, false);
}

// A-fragment dword index (16-bit A 16x32): pair base K = 2v + 16*half
__device__ __forceinline__ int a_kpair(int v, int hf) {
  return v + (hf << 3);
}

// 4 consecutive 16B chunks: global -> LDS, tracked by ASYNCcnt.
// ISA: INST_OFFSET is added to BOTH the LDS and the global address.
__device__ __forceinline__ void async4_b128(unsigned lds_byte, const void* base,
                                            unsigned gbyte) {
  asm volatile(
      "global_load_async_to_lds_b128 %0, %1, %2\n\t"
      "global_load_async_to_lds_b128 %0, %1, %2 offset:16\n\t"
      "global_load_async_to_lds_b128 %0, %1, %2 offset:32\n\t"
      "global_load_async_to_lds_b128 %0, %1, %2 offset:48"
      :
      : "v"(lds_byte), "v"(gbyte), "s"(base)
      : "memory");
}

__device__ __forceinline__ void wait_async0() {
  asm volatile("s_wait_asynccnt 0x0" ::: "memory");
}

// ---------------------------------------------------------------------------
// Pre-pass: f32 -> bf16 elementwise (q), and 32x32 LDS-tiled transpose+convert
// for weights: Wt[n][k] = bf16(W[k][n]).
// ---------------------------------------------------------------------------
__global__ __launch_bounds__(256) void cvt_bf16(const float* __restrict__ in,
                                                unsigned short* __restrict__ out) {
  int i = (blockIdx.x * 256 + threadIdx.x) * 4;
  float4 f = *(const float4*)(in + i);
  uint2 u;
  u.x = (unsigned)f2bf(f.x) | ((unsigned)f2bf(f.y) << 16);
  u.y = (unsigned)f2bf(f.z) | ((unsigned)f2bf(f.w) << 16);
  *(uint2*)(out + i) = u;
}

__global__ __launch_bounds__(256) void transpose_bf16(
    const float* __restrict__ W, unsigned short* __restrict__ Wt) {
  __shared__ float t[32][33];
  const int k0 = blockIdx.x * 32, n0 = blockIdx.y * 32;
  const int tid = threadIdx.x;
#pragma unroll
  for (int i = 0; i < 4; ++i) {
    int idx = tid + i * 256;
    int r = idx >> 5, c = idx & 31;
    t[r][c] = W[(size_t)(k0 + r) * 1024 + n0 + c];
  }
  __syncthreads();
#pragma unroll
  for (int i = 0; i < 4; ++i) {
    int idx = tid + i * 256;
    int r = idx >> 5, c = idx & 31;
    Wt[(size_t)(n0 + r) * 1024 + k0 + c] = f2bf(t[c][r]);
  }
}

// ---------------------------------------------------------------------------
// 128x128 GEMM, K-step 64, double-buffered async-DMA LDS tiles.
// A: bf16 [4096][1024] row-major.  Bt: bf16 [1024 n][1024 k].
// MODE 0: out bf16 head-split [b,h,s,d]              (K projection)
// MODE 1: out bf16 head-split [b,h,s,d], x 1/sqrt(d) (Q projection)
// MODE 2: out bf16 head-split-T [b,h,d,s]            (V projection)
// MODE 3: out f32 row-major + bias                   (output projection)
// ---------------------------------------------------------------------------
constexpr int SAB = 72;  // LDS row stride in bf16: 144B (16B-aligned, conflict-free)

template <int MODE>
__global__ __launch_bounds__(256) void gemm_async(
    const unsigned short* __restrict__ A, const unsigned short* __restrict__ Bt,
    const float* __restrict__ bias, unsigned short* __restrict__ Ob,
    float* __restrict__ Of) {
  __shared__ __align__(16) unsigned short As[2][128 * SAB];
  __shared__ __align__(16) unsigned short Bs[2][128 * SAB];

  const int tid = threadIdx.x;
  const int wave = tid >> 5, lane = tid & 31;
  const int hf = lane >> 4, ln = lane & 15;
  const int wm = wave & 1, wn = wave >> 1;
  const int m0 = blockIdx.y * 128, n0 = blockIdx.x * 128;

  // DMA addressing: each thread fills 64B (4 x b128) of one 128B tile row.
  const int arow = tid >> 1;
  const int acol = (tid & 1) * 64;

  auto issue = [&](int k0, int buf) {
    unsigned la = (unsigned)(uintptr_t)&As[buf][arow * SAB] + acol;
    unsigned ga = (unsigned)(((m0 + arow) * 1024 + k0) * 2) + acol;
    async4_b128(la, A, ga);
    unsigned lb = (unsigned)(uintptr_t)&Bs[buf][arow * SAB] + acol;
    unsigned gb = (unsigned)(((n0 + arow) * 1024 + k0) * 2) + acol;
    async4_b128(lb, Bt, gb);
  };

  v8f acc[4][2];
#pragma unroll
  for (int mt = 0; mt < 4; ++mt)
#pragma unroll
    for (int nt = 0; nt < 2; ++nt) acc[mt][nt] = vzero8();

  issue(0, 0);
  for (int t = 0; t < 16; ++t) {
    wait_async0();
    __syncthreads();  // tile t resident; all waves done with tile t-1 compute
    if (t + 1 < 16) issue((t + 1) * 64, (t + 1) & 1);

    const unsigned short* as = As[t & 1];
    const unsigned short* bs = Bs[t & 1];
#pragma unroll
    for (int ks = 0; ks < 2; ++ks) {
      Frag16 a[4];
#pragma unroll
      for (int mt = 0; mt < 4; ++mt) {
        const unsigned int* src =
            (const unsigned int*)&as[(wm * 64 + mt * 16 + ln) * SAB];
#pragma unroll
        for (int v = 0; v < 8; ++v) a[mt].u[v] = src[ks * 16 + a_kpair(v, hf)];
      }
      Frag16 b[2];
#pragma unroll
      for (int nt = 0; nt < 2; ++nt) {
        const uint4* src =
            (const uint4*)&bs[(wn * 32 + nt * 16 + ln) * SAB + ks * 32 + hf * 16];
        b[nt].q[0] = src[0];
        b[nt].q[1] = src[1];
      }
#pragma unroll
      for (int mt = 0; mt < 4; ++mt)
#pragma unroll
        for (int nt = 0; nt < 2; ++nt)
          acc[mt][nt] = wmma_bf16(a[mt].v, b[nt].v, acc[mt][nt]);
    }
    __syncthreads();
  }

  // ---- epilogue ----
#pragma unroll
  for (int mt = 0; mt < 4; ++mt) {
#pragma unroll
    for (int nt = 0; nt < 2; ++nt) {
      int ng = n0 + wn * 32 + nt * 16 + ln;
      float bv = 0.0f;
      if constexpr (MODE == 3) bv = bias[ng];
#pragma unroll
      for (int v = 0; v < 8; ++v) {
        int m = m0 + wm * 64 + mt * 16 + v + 8 * hf;
        float val = acc[mt][nt][v];
        if constexpr (MODE == 3) {
          Of[(size_t)m * 1024 + ng] = val + bv;
        } else {
          if constexpr (MODE == 1) val *= 0.125f;  // fold 1/sqrt(64) into Q
          int bb = m >> 11, s = m & 2047;
          int h = ng >> 6, d = ng & 63;
          size_t idx;
          if constexpr (MODE == 2)
            idx = ((size_t)(bb * 16 + h) * 64 + d) * 2048 + s;  // [b,h,d,s]
          else
            idx = ((size_t)(bb * 16 + h) * 2048 + s) * 64 + d;  // [b,h,s,d]
          Ob[idx] = f2bf(val);
        }
      }
    }
  }
}

// ---------------------------------------------------------------------------
// Flash attention: grid (32 seq-tiles, 32 b*h), 4 waves/block, 16 q-rows/wave.
// Q arrives pre-scaled, so scores come out of WMMA ready for softmax.
// ---------------------------------------------------------------------------
__global__ __launch_bounds__(128) void attn64(
    const unsigned short* __restrict__ Qb, const unsigned short* __restrict__ Kb,
    const unsigned short* __restrict__ Vt, unsigned short* __restrict__ Cx) {
  constexpr int SEQ = 2048, D = 64, SP = 68;
  __shared__ __align__(16) unsigned short Plds[4][16 * SP];

  const int bh = blockIdx.y;
  const int wave = threadIdx.x >> 5, lane = threadIdx.x & 31;
  const int hf = lane >> 4, ln = lane & 15;
  const int q0 = blockIdx.x * 64 + wave * 16;
  const size_t hb = (size_t)bh * SEQ * D;

  Frag16 qa[2];
  {
    const unsigned int* qrow =
        (const unsigned int*)(Qb + hb + (size_t)(q0 + ln) * D);
#pragma unroll
    for (int ks = 0; ks < 2; ++ks)
#pragma unroll
      for (int v = 0; v < 8; ++v) qa[ks].u[v] = qrow[ks * 16 + a_kpair(v, hf)];
  }

  v8f acc[4];
#pragma unroll
  for (int nt = 0; nt < 4; ++nt) acc[nt] = vzero8();
  v8f rmax, rsum;
#pragma unroll
  for (int v = 0; v < 8; ++v) { rmax[v] = -3.0e38f; rsum[v] = 0.0f; }

  unsigned short* pl = Plds[wave];
  const unsigned int* plu = (const unsigned int*)pl;

  for (int kt = 0; kt < 32; ++kt) {
    const int nb = kt * 64;

    v8f S[4];
#pragma unroll
    for (int nt = 0; nt < 4; ++nt) {
      const uint4* krow =
          (const uint4*)(Kb + hb + (size_t)(nb + nt * 16 + ln) * D);
      Frag16 kb0, kb1;
      kb0.q[0] = krow[hf * 2 + 0];
      kb0.q[1] = krow[hf * 2 + 1];
      kb1.q[0] = krow[4 + hf * 2 + 0];
      kb1.q[1] = krow[4 + hf * 2 + 1];
      v8f s = wmma_bf16(qa[0].v, kb0.v, vzero8());
      s = wmma_bf16(qa[1].v, kb1.v, s);
      S[nt] = s;
    }

    v8f tmax;
#pragma unroll
    for (int v = 0; v < 8; ++v)
      tmax[v] = fmaxf(fmaxf(S[0][v], S[1][v]), fmaxf(S[2][v], S[3][v]));
#pragma unroll
    for (int mask = 1; mask <= 8; mask <<= 1)
#pragma unroll
      for (int v = 0; v < 8; ++v)
        tmax[v] = fmaxf(tmax[v], __shfl_xor(tmax[v], mask, 32));

    v8f corr, tsum;
#pragma unroll
    for (int v = 0; v < 8; ++v) {
      float nm = fmaxf(rmax[v], tmax[v]);
      corr[v] = __expf(rmax[v] - nm);
      rmax[v] = nm;
      tsum[v] = 0.0f;
    }
#pragma unroll
    for (int nt = 0; nt < 4; ++nt)
#pragma unroll
      for (int v = 0; v < 8; ++v) {
        float p = __expf(S[nt][v] - rmax[v]);
        S[nt][v] = p;
        tsum[v] += p;
      }
#pragma unroll
    for (int mask = 1; mask <= 8; mask <<= 1)
#pragma unroll
      for (int v = 0; v < 8; ++v) tsum[v] += __shfl_xor(tsum[v], mask, 32);
#pragma unroll
    for (int v = 0; v < 8; ++v) rsum[v] = rsum[v] * corr[v] + tsum[v];
#pragma unroll
    for (int nt = 0; nt < 4; ++nt)
#pragma unroll
      for (int v = 0; v < 8; ++v) acc[nt][v] *= corr[v];

    // transpose P: C-layout -> A-layout through per-wave LDS (in-order DS ops)
#pragma unroll
    for (int nt = 0; nt < 4; ++nt)
#pragma unroll
      for (int v = 0; v < 8; ++v)
        pl[(v + 8 * hf) * SP + nt * 16 + ln] = f2bf(S[nt][v]);

    Frag16 pa[2];
#pragma unroll
    for (int ks = 0; ks < 2; ++ks)
#pragma unroll
      for (int v = 0; v < 8; ++v)
        pa[ks].u[v] = plu[ln * (SP / 2) + ks * 16 + a_kpair(v, hf)];

#pragma unroll
    for (int nt = 0; nt < 4; ++nt) {
      const uint4* vrow =
          (const uint4*)(Vt + hb + (size_t)(nt * 16 + ln) * SEQ + nb);
      Frag16 vb0, vb1;
      vb0.q[0] = vrow[hf * 2 + 0];
      vb0.q[1] = vrow[hf * 2 + 1];
      vb1.q[0] = vrow[4 + hf * 2 + 0];
      vb1.q[1] = vrow[4 + hf * 2 + 1];
      acc[nt] = wmma_bf16(pa[0].v, vb0.v, acc[nt]);
      acc[nt] = wmma_bf16(pa[1].v, vb1.v, acc[nt]);
    }
  }

  const int bb = bh >> 4, h = bh & 15;
#pragma unroll
  for (int v = 0; v < 8; ++v) {
    float inv = 1.0f / rsum[v];
    int m = q0 + v + 8 * hf;
#pragma unroll
    for (int nt = 0; nt < 4; ++nt) {
      int col = h * 64 + nt * 16 + ln;
      Cx[((size_t)(bb * 2048 + m)) * 1024 + col] = f2bf(acc[nt][v] * inv);
    }
  }
}

// ---------------------------------------------------------------------------
extern "C" void kernel_launch(void* const* d_in, const int* in_sizes, int n_in,
                              void* d_out, int out_size, void* d_ws,
                              size_t ws_size, hipStream_t stream) {
  (void)in_sizes; (void)n_in; (void)out_size; (void)ws_size;
  const float* q  = (const float*)d_in[0];
  const float* Wq = (const float*)d_in[1];
  const float* Wk = (const float*)d_in[2];
  const float* Wv = (const float*)d_in[3];
  const float* Wo = (const float*)d_in[4];
  const float* bo = (const float*)d_in[5];
  float* out = (float*)d_out;

  const size_t QE = (size_t)4096 * 1024;  // 4M elems
  const size_t WE = (size_t)1024 * 1024;  // 1M elems
  unsigned short* qb  = (unsigned short*)d_ws;
  unsigned short* WqT = qb + QE;
  unsigned short* WkT = WqT + WE;
  unsigned short* WvT = WkT + WE;
  unsigned short* WoT = WvT + WE;
  unsigned short* Qh  = WoT + WE;
  unsigned short* Kh  = Qh + QE;
  unsigned short* Vt  = Kh + QE;
  unsigned short* Cx  = Vt + QE;

  // pre-pass: bf16 conversion + weight transposes
  cvt_bf16<<<4096, 256, 0, stream>>>(q, qb);
  dim3 tg(32, 32);
  transpose_bf16<<<tg, 256, 0, stream>>>(Wq, WqT);
  transpose_bf16<<<tg, 256, 0, stream>>>(Wk, WkT);
  transpose_bf16<<<tg, 256, 0, stream>>>(Wv, WvT);
  transpose_bf16<<<tg, 256, 0, stream>>>(Wo, WoT);

  dim3 gg(1024 / 128, 4096 / 128);  // (N tiles, M tiles)
  gemm_async<1><<<gg, 256, 0, stream>>>(qb, WqT, nullptr, Qh, nullptr);
  gemm_async<0><<<gg, 256, 0, stream>>>(qb, WkT, nullptr, Kh, nullptr);
  gemm_async<2><<<gg, 256, 0, stream>>>(qb, WvT, nullptr, Vt, nullptr);
  attn64<<<dim3(32, 32), 128, 0, stream>>>(Qh, Kh, Vt, Cx);
  gemm_async<3><<<gg, 256, 0, stream>>>(Cx, WoT, bo, nullptr, out);
}
